// SelfAttention_30640296690306
// MI455X (gfx1250) — matrix operand order repaired and checked by reading, not compiled, not run on hardware
//
#include <hip/hip_runtime.h>
#include <stdint.h>

// ---------------------------------------------------------------------------
// MHA forward for MI455X (gfx1250, wave32, WMMA).
//   Q/K/V/O projections + flash attention, all matmuls on
//   v_wmma_f32_16x16x32_bf16 (bf16 inputs, f32 accumulate).
// ---------------------------------------------------------------------------

#define D_MODEL 1024
#define N_HEADS 16
#define D_H     64
#define B_SZ    4
#define T_SEQ   2048
#define BT      (B_SZ * T_SEQ)

typedef __attribute__((ext_vector_type(16))) __bf16       v16bf;
typedef __attribute__((ext_vector_type(8)))  float        v8f;
typedef __attribute__((ext_vector_type(4)))  unsigned int u32x4;
typedef __attribute__((ext_vector_type(4)))  float        f32x4;
typedef unsigned short ushort_t;
typedef unsigned int   u32;

// f32 -> bf16 round-to-nearest-even
__device__ __forceinline__ ushort_t f2bf(float f) {
    u32 x = __float_as_uint(f);
    u32 r = x + 0x7FFFu + ((x >> 16) & 1u);
    return (ushort_t)(r >> 16);
}

// pack 8 consecutive f32 -> 8 bf16 in one u32x4
__device__ __forceinline__ u32x4 pack8(const float* p) {
    f32x4 f0 = *(const f32x4*)p;
    f32x4 f1 = *(const f32x4*)(p + 4);
    u32x4 pk;
    pk[0] = (u32)f2bf(f0[0]) | ((u32)f2bf(f0[1]) << 16);
    pk[1] = (u32)f2bf(f0[2]) | ((u32)f2bf(f0[3]) << 16);
    pk[2] = (u32)f2bf(f1[0]) | ((u32)f2bf(f1[1]) << 16);
    pk[3] = (u32)f2bf(f1[2]) | ((u32)f2bf(f1[3]) << 16);
    return pk;
}

union BOp { v16bf v; u32x4 q[2]; };   // one WMMA A/B operand (16 bf16 = 8 VGPRs)

// ---------------------------------------------------------------------------
// GEMM: y = in * W^T + bias.   in: [BT][1024] (f32 for MODE 0/1, bf16 MODE 2)
// MODE 0: out bf16 (B,H,T,64)   (Q, K)
// MODE 1: out bf16 (B,H,64,T)   (V transposed, for contiguous P*V B-operands)
// MODE 2: out f32  (B,T,1024)   (final projection -> d_out)
// Block: 256 thr (8 waves), tile 64x128, K-step 64.
// Wave = 16 rows x 64 cols: 4 f32 accumulators, A reused over 4 B tiles,
// 8 WMMAs per stage between barriers.
// ---------------------------------------------------------------------------
template <int MODE>
__global__ __launch_bounds__(256)
void gemm_proj(const void* __restrict__ inp, const float* __restrict__ W,
               const float* __restrict__ bias, void* __restrict__ out) {
    __shared__ __align__(16) ushort_t Xs[64][72];    // 64 rows x 64 k (+8 pad)
    __shared__ __align__(16) ushort_t Ws[128][72];   // 128 cols x 64 k (+8 pad)

    const int tid  = threadIdx.x;
    const int lane = tid & 31;
    const int wave = tid >> 5;
    const int rw   = wave >> 1;          // row group 0..3  (16 rows each)
    const int cw   = wave & 1;           // col group 0..1  (64 cols each)
    const int row0 = blockIdx.x * 64;
    const int col0 = blockIdx.y * 128;

    const int m    = lane & 15;
    const int hi   = lane >> 4;
    const int ch   = hi * 8;             // A-operand k-offset within half
    const int koff = hi * 16;            // unused; B operand also uses ch

    const int sr = tid >> 2;             // staging row 0..63
    const int sc = (tid & 3) * 16;       // staging col 0,16,32,48

    v8f acc[4] = {v8f{}, v8f{}, v8f{}, v8f{}};

    for (int k0 = 0; k0 < D_MODEL; k0 += 64) {
        // ---- stage input tile (64 rows x 64 k) as bf16 ----
        if (MODE == 2) {
            const ushort_t* xp = (const ushort_t*)inp + (size_t)(row0 + sr) * D_MODEL + k0 + sc;
            *(u32x4*)&Xs[sr][sc]     = *(const u32x4*)xp;
            *(u32x4*)&Xs[sr][sc + 8] = *(const u32x4*)(xp + 8);
        } else {
            const float* xp = (const float*)inp + (size_t)(row0 + sr) * D_MODEL + k0 + sc;
            *(u32x4*)&Xs[sr][sc]     = pack8(xp);
            *(u32x4*)&Xs[sr][sc + 8] = pack8(xp + 8);
        }
        // ---- stage weight tile Ws[n][k] = W[col0+n][k0+k], 128 rows ----
#pragma unroll
        for (int rep = 0; rep < 2; ++rep) {
            const int wr = rep * 64 + sr;
            const float* wp = W + (size_t)(col0 + wr) * D_MODEL + k0 + sc;
            *(u32x4*)&Ws[wr][sc]     = pack8(wp);
            *(u32x4*)&Ws[wr][sc + 8] = pack8(wp + 8);
        }
        __syncthreads();

#pragma unroll
        for (int kh = 0; kh < 2; ++kh) {             // two K-32 halves
            // A operand: row m, K = {0..7,16..23} / {8..15,24..31}
            const ushort_t* ar = &Xs[rw * 16 + m][kh * 32];
            BOp a;
            a.q[0] = *(const u32x4*)(ar + ch);
            a.q[1] = *(const u32x4*)(ar + ch + 16);
#pragma unroll
            for (int nt = 0; nt < 4; ++nt) {         // 4 N tiles, A reused
                const ushort_t* br = &Ws[cw * 64 + nt * 16 + m][kh * 32 + ch];
                BOp bb;
                bb.q[0] = *(const u32x4*)br;
                bb.q[1] = *(const u32x4*)(br + 16);
                acc[nt] = __builtin_amdgcn_wmma_f32_16x16x32_bf16(
                    false, a.v, false, bb.v, (short)0, acc[nt], false, false);
            }
        }
        __syncthreads();
    }

    // ---- epilogue: bias add + store (C layout: row = i + 8*hi, col = lane&15)
#pragma unroll
    for (int nt = 0; nt < 4; ++nt) {
        const int ncol = col0 + cw * 64 + nt * 16 + m;
        const float bval = bias[ncol];
        const int h  = ncol >> 6;
        const int dh = ncol & 63;
#pragma unroll
        for (int i = 0; i < 8; ++i) {
            const int r  = row0 + rw * 16 + i + hi * 8;   // global row (b*T + t)
            const int bb = r / T_SEQ;
            const int tt = r % T_SEQ;
            const float val = acc[nt][i] + bval;
            if (MODE == 0) {
                ((ushort_t*)out)[(((size_t)(bb * N_HEADS + h) * T_SEQ) + tt) * D_H + dh] = f2bf(val);
            } else if (MODE == 1) {
                ((ushort_t*)out)[(((size_t)(bb * N_HEADS + h) * D_H) + dh) * T_SEQ + tt] = f2bf(val);
            } else {
                ((float*)out)[(size_t)r * D_MODEL + ncol] = val;
            }
        }
    }
}

// ---------------------------------------------------------------------------
// Flash attention: block = (b, h, 64-query tile), 4 waves x 16 queries.
// Q (B,H,T,64) bf16, K (B,H,T,64) bf16, Vt (B,H,64,T) bf16 -> Y (B,T,1024) bf16
// ---------------------------------------------------------------------------
__global__ __launch_bounds__(128)
void attn_kernel(const ushort_t* __restrict__ Q, const ushort_t* __restrict__ K,
                 const ushort_t* __restrict__ Vt, const unsigned char* __restrict__ pad,
                 ushort_t* __restrict__ Y) {
    __shared__ __align__(16) ushort_t Plds[4][16][32];   // per-wave P transpose buffer

    const int lane = threadIdx.x & 31;
    const int wave = threadIdx.x >> 5;
    const int b = blockIdx.z, h = blockIdx.y;
    const int q0 = blockIdx.x * 64 + wave * 16;

    const size_t bh = (size_t)(b * N_HEADS + h);
    const ushort_t* Qb = Q  + bh * T_SEQ * D_H;
    const ushort_t* Kb = K  + bh * T_SEQ * D_H;
    const ushort_t* Vb = Vt + bh * D_H * T_SEQ;

    const int m  = lane & 15;
    const int hi = lane >> 4;
    const int ch = hi * 8;

    // Q A-operands, resident for the whole loop (d_h halves 0..31 / 32..63)
    const ushort_t* qptr = Qb + (size_t)(q0 + m) * D_H;
    BOp aq0, aq1;
    aq0.q[0] = *(const u32x4*)(qptr + ch);
    aq0.q[1] = *(const u32x4*)(qptr + ch + 16);
    aq1.q[0] = *(const u32x4*)(qptr + 32 + ch);
    aq1.q[1] = *(const u32x4*)(qptr + 32 + ch + 16);

    v8f o[4] = {v8f{}, v8f{}, v8f{}, v8f{}};
    float mrun[8], lrun[8];
#pragma unroll
    for (int i = 0; i < 8; ++i) { mrun[i] = -3.0e38f; lrun[i] = 0.0f; }
    const float NEGBIG = -3.0e38f;

    const int kend = q0 + 16;
    for (int kc = 0; kc < kend; kc += 32) {         // 32 keys per chunk
        // prefetch next chunk's K rows / V columns while we compute this one
        if (kc + 32 < kend) {
            __builtin_prefetch((const void*)(Kb + (size_t)(kc + 32 + m) * D_H), 0, 1);
            __builtin_prefetch((const void*)(Vb + (size_t)m * T_SEQ + kc + 32), 0, 1);
        }

        // ---- S = (Q K^T) / 8 : two 16x16 tiles, each K-chained over d_h ----
        v8f s[2] = {v8f{}, v8f{}};
#pragma unroll
        for (int t = 0; t < 2; ++t) {
            const ushort_t* krow = Kb + (size_t)(kc + t * 16 + m) * D_H;
            const int koff = hi * 8;
            BOp bk0, bk1;
            bk0.q[0] = *(const u32x4*)(krow + koff);
            bk0.q[1] = *(const u32x4*)(krow + koff + 16);
            bk1.q[0] = *(const u32x4*)(krow + 32 + koff);
            bk1.q[1] = *(const u32x4*)(krow + 32 + koff + 16);
            s[t] = __builtin_amdgcn_wmma_f32_16x16x32_bf16(false, aq0.v, false, bk0.v, (short)0, s[t], false, false);
            s[t] = __builtin_amdgcn_wmma_f32_16x16x32_bf16(false, aq1.v, false, bk1.v, (short)0, s[t], false, false);
        }

        // ---- mask + online softmax (rows split across VGPR idx / lane half) --
        const unsigned char pm0 = pad[b * T_SEQ + kc + m];
        const unsigned char pm1 = pad[b * T_SEQ + kc + 16 + m];
        float rmax[8];
#pragma unroll
        for (int i = 0; i < 8; ++i) {
            const int qi = q0 + i + hi * 8;
            float s0 = s[0][i] * 0.125f;
            float s1 = s[1][i] * 0.125f;
            if ((kc + m > qi) || pm0)      s0 = NEGBIG;
            if ((kc + 16 + m > qi) || pm1) s1 = NEGBIG;
            s[0][i] = s0; s[1][i] = s1;
            float t = fmaxf(s0, s1);
            t = fmaxf(t, __shfl_xor(t, 1, 32));
            t = fmaxf(t, __shfl_xor(t, 2, 32));
            t = fmaxf(t, __shfl_xor(t, 4, 32));
            t = fmaxf(t, __shfl_xor(t, 8, 32));
            rmax[i] = t;
        }
#pragma unroll
        for (int i = 0; i < 8; ++i) {
            const float mnew = fmaxf(mrun[i], rmax[i]);
            const float corr = __expf(mrun[i] - mnew);
            const float p0 = __expf(s[0][i] - mnew);
            const float p1 = __expf(s[1][i] - mnew);
            float rs = p0 + p1;
            rs += __shfl_xor(rs, 1, 32);
            rs += __shfl_xor(rs, 2, 32);
            rs += __shfl_xor(rs, 4, 32);
            rs += __shfl_xor(rs, 8, 32);
            lrun[i] = lrun[i] * corr + rs;
            mrun[i] = mnew;
#pragma unroll
            for (int c = 0; c < 4; ++c) o[c][i] *= corr;
            Plds[wave][i + hi * 8][m]      = f2bf(p0);   // C-layout -> plain [16][32]
            Plds[wave][i + hi * 8][m + 16] = f2bf(p1);
        }
        __asm__ volatile("s_wait_dscnt 0x0" ::: "memory");

        // ---- P back as A-operand (16x32), V^T gives contiguous B-operands ----
        BOp ap;
        ap.q[0] = *(const u32x4*)&Plds[wave][m][ch];
        ap.q[1] = *(const u32x4*)&Plds[wave][m][ch + 16];
#pragma unroll
        for (int c = 0; c < 4; ++c) {
            const ushort_t* vrow = Vb + (size_t)(c * 16 + m) * T_SEQ + kc + hi * 8;
            BOp bvv;
            bvv.q[0] = *(const u32x4*)vrow;
            bvv.q[1] = *(const u32x4*)(vrow + 16);
            o[c] = __builtin_amdgcn_wmma_f32_16x16x32_bf16(false, ap.v, false, bvv.v, (short)0, o[c], false, false);
        }
    }

    // ---- normalize and store Y (B,T,1024) bf16 ----
#pragma unroll
    for (int i = 0; i < 8; ++i) {
        const float inv = 1.0f / lrun[i];
        const int t = q0 + i + hi * 8;
#pragma unroll
        for (int c = 0; c < 4; ++c) {
            Y[((size_t)b * T_SEQ + t) * D_MODEL + h * D_H + c * 16 + m] = f2bf(o[c][i] * inv);
        }
    }
}

// ---------------------------------------------------------------------------
extern "C" void kernel_launch(void* const* d_in, const int* in_sizes, int n_in,
                              void* d_out, int out_size, void* d_ws, size_t ws_size,
                              hipStream_t stream) {
    const float*         x   = (const float*)d_in[0];
    const unsigned char* pad = (const unsigned char*)d_in[1];
    const float* Wq = (const float*)d_in[2];
    const float* bq = (const float*)d_in[3];
    const float* Wk = (const float*)d_in[4];
    const float* bk = (const float*)d_in[5];
    const float* Wv = (const float*)d_in[6];
    const float* bv = (const float*)d_in[7];
    const float* Wo = (const float*)d_in[8];
    const float* bo = (const float*)d_in[9];

    // workspace: Q,K (B,H,T,64) | V^T (B,H,64,T) | Y (B,T,1024), all bf16 -> 64 MB
    ushort_t* Qw = (ushort_t*)d_ws;
    ushort_t* Kw = Qw + (size_t)BT * D_MODEL;
    ushort_t* Vw = Kw + (size_t)BT * D_MODEL;
    ushort_t* Yw = Vw + (size_t)BT * D_MODEL;

    dim3 gblk(BT / 64, D_MODEL / 128);
    gemm_proj<0><<<gblk, 256, 0, stream>>>(x,  Wq, bq, Qw);
    gemm_proj<0><<<gblk, 256, 0, stream>>>(x,  Wk, bk, Kw);
    gemm_proj<1><<<gblk, 256, 0, stream>>>(x,  Wv, bv, Vw);

    dim3 ablk(T_SEQ / 64, N_HEADS, B_SZ);
    attn_kernel<<<ablk, 128, 0, stream>>>(Qw, Kw, Vw, pad, Yw);

    gemm_proj<2><<<gblk, 256, 0, stream>>>(Yw, Wo, bo, d_out);
}
